// Qwen3MoEDecoderLayer_12893491823394
// MI455X (gfx1250) — matrix operand order repaired and checked by reading, not compiled, not run on hardware
//
#include <hip/hip_runtime.h>
#include <cstddef>

// ---------------------------------------------------------------------------
// Qwen3-MoE decoder layer for MI455X (gfx1250, wave32, WMMA bf16).
// B=1, S=1024, H=2048, NH=32, NKV=4, HD=128, E=32, topK=8, I=768, fp32 I/O.
// Memory-bound on fp32 weight traffic (~670MB @ 23.3TB/s); bf16 WMMA with
// v_cvt_pk_bf16_f32 staging in LDS, fragments stored K-linear, contiguous k order
// (2x ds_load_b128 per operand), permlane16 cross-lane reductions (no LDS).
// ---------------------------------------------------------------------------

constexpr int T_    = 1024;
constexpr int H_    = 2048;
constexpr int NH_   = 32;
constexpr int NKV_  = 4;
constexpr int HD_   = 128;
constexpr int E_    = 32;
constexpr int TOPK_ = 8;
constexpr int I_    = 768;
constexpr float EPS_ = 1e-6f;

typedef __attribute__((ext_vector_type(16))) __bf16        v16bf;
typedef __attribute__((ext_vector_type(4)))  __bf16        v4bf;
typedef __attribute__((ext_vector_type(8)))  float         v8f;
typedef __attribute__((ext_vector_type(4)))  float         v4f;
typedef __attribute__((ext_vector_type(4)))  unsigned int  uint4v;

union FragBF { v16bf v; uint4v q[2]; unsigned short u[16]; };

// fp32 -> bf16 via native convert (v_cvt_pk_bf16_f32).
__device__ __forceinline__ unsigned short f2bf(float f) {
  union { __bf16 b; unsigned short u; } c;
  c.b = (__bf16)f;
  return c.u;
}

__device__ __forceinline__ unsigned long long pack4bf(float a, float b,
                                                      float c, float d) {
  v4f x = {a, b, c, d};
  v4bf y = __builtin_convertvector(x, v4bf);
  union { v4bf v; unsigned long long u; } cv;
  cv.v = y;
  return cv.u;
}

// A-fragment K interleave (ISA 7.12.2): half j of lane l holds
// K = (l<16?0:8) + (j<8 ? j : j+8).  permk maps logical K -> physical slot so
// each lane's 16 halves are contiguous (lanes<16: slots 0..15, else 16..31).
__device__ __forceinline__ int permk(int k) {
  return (k & 7) | ((k & 8) << 1) | ((k & 16) >> 1);
}

__device__ __forceinline__ void lds_fence() {
  asm volatile("s_wait_dscnt 0" ::: "memory");
}

// --- VALU cross-lane reductions via v_permlane16_b32 (no LDS/ds_bpermute) ---
// Selector nibble j = source lane for lane j (within a 16-lane row).
__device__ __forceinline__ float pl16(float v, unsigned s0, unsigned s1) {
  union { float f; unsigned u; } a, r;
  a.f = v;
  r.u = __builtin_amdgcn_permlane16(a.u, a.u, s0, s1, false, false);
  return r.f;
}
__device__ __forceinline__ float plx16(float v) {   // swap 16-lane rows
  union { float f; unsigned u; } a, r;
  a.f = v;
  r.u = __builtin_amdgcn_permlanex16(a.u, a.u, 0x76543210u, 0xFEDCBA98u,
                                     false, false);
  return r.f;
}

__device__ __forceinline__ float half_reduce_sum(float v) {
  v += pl16(v, 0x67452301u, 0xEFCDAB89u);   // xor 1
  v += pl16(v, 0x54761032u, 0xDCFE98BAu);   // xor 2
  v += pl16(v, 0x32107654u, 0xBA98FEDCu);   // xor 4
  v += pl16(v, 0xFEDCBA98u, 0x76543210u);   // xor 8
  return v;
}
__device__ __forceinline__ float half_reduce_max(float v) {
  v = fmaxf(v, pl16(v, 0x67452301u, 0xEFCDAB89u));
  v = fmaxf(v, pl16(v, 0x54761032u, 0xDCFE98BAu));
  v = fmaxf(v, pl16(v, 0x32107654u, 0xBA98FEDCu));
  v = fmaxf(v, pl16(v, 0xFEDCBA98u, 0x76543210u));
  return v;
}
__device__ __forceinline__ float wave_reduce_sum(float v) {
  v = half_reduce_sum(v);
  v += plx16(v);
  return v;
}
__device__ __forceinline__ float wave_reduce_max(float v) {
  v = half_reduce_max(v);
  v = fmaxf(v, plx16(v));
  return v;
}

// ---------------------------------------------------------------------------
// RMSNorm over rows of [T_, H_]
// ---------------------------------------------------------------------------
__global__ void __launch_bounds__(256)
rmsnorm_kernel(const float* __restrict__ x, const float* __restrict__ w,
               float* __restrict__ out) {
  __shared__ float red[8];
  const int row = blockIdx.x;
  const int tid = threadIdx.x;
  const float* xr = x + (size_t)row * H_;
  float ss = 0.f;
#pragma unroll
  for (int i = 0; i < H_ / 256; ++i) { float v = xr[tid + i * 256]; ss += v * v; }
  ss = wave_reduce_sum(ss);
  if ((tid & 31) == 0) red[tid >> 5] = ss;
  __syncthreads();
  float tot = 0.f;
#pragma unroll
  for (int i = 0; i < 8; ++i) tot += red[i];
  const float r = rsqrtf(tot / (float)H_ + EPS_);
  float* orow = out + (size_t)row * H_;
#pragma unroll
  for (int i = 0; i < H_ / 256; ++i) {
    int idx = tid + i * 256;
    orow[idx] = xr[idx] * r * w[idx];
  }
}

// ---------------------------------------------------------------------------
// bf16-WMMA GEMM: C[M,N] = A[M,K]*B[K,N]  (fp32 in/out)
//   MODE 0: C = acc ; MODE 1: C = acc + R ; MODE 2: C += rowscale[row]*acc
// 256 thr = 8 waves, tile 64x128, K-step 64, 8 WMMAs per wave per barrier.
// ---------------------------------------------------------------------------
#define BM 64
#define BN 128
#define BK 64
#define PAD 8

template <int MODE>
__global__ void __launch_bounds__(256)
gemm_bf16_kernel(const float* __restrict__ A, const float* __restrict__ B,
                 float* __restrict__ C, const float* __restrict__ R,
                 const float* __restrict__ rowscale, int rs_stride,
                 int M, int N, int Kdim) {
  __shared__ unsigned short As[BM][BK + PAD];    // row-major, K linear
  __shared__ unsigned short BsT[BN][BK + PAD];   // transposed (n-major), K linear

  const int row0 = blockIdx.y * BM;
  const int col0 = blockIdx.x * BN;
  const int tid  = threadIdx.x;
  const int wave = tid >> 5;
  const int lane = tid & 31;
  const int wm   = (wave & 3) * 16;
  const int wn   = (wave >> 2) * 64;
  const int m    = lane & 15;
  const int hsel = (lane < 16) ? 0 : 16;   // physical half-run select

  v8f acc[4];
#pragma unroll
  for (int t = 0; t < 4; ++t) acc[t] = (v8f){0, 0, 0, 0, 0, 0, 0, 0};

  for (int k0 = 0; k0 < Kdim; k0 += BK) {
    // A: 64x64 fp32, float4 loads -> K-linear bf16 b64 stores
#pragma unroll
    for (int g = 0; g < 4; ++g) {
      int e = (tid + g * 256) * 4;
      int r = e >> 6, c = e & 63;
      float4 v = *reinterpret_cast<const float4*>(
          A + (size_t)(row0 + r) * Kdim + k0 + c);
      int cp = c;
      *reinterpret_cast<unsigned long long*>(&As[r][cp]) =
          pack4bf(v.x, v.y, v.z, v.w);
    }
    // B: 64x128 fp32 -> transposed bf16 (each thread: fixed n, 4 consecutive k)
#pragma unroll
    for (int g = 0; g < 8; ++g) {
      int gg = tid + g * 256;
      int nn = gg & 127;
      int kq = (gg >> 7) * 4;
      const float* bp = B + (size_t)(k0 + kq) * N + col0 + nn;
      float b0 = bp[0];
      float b1 = bp[(size_t)N];
      float b2 = bp[(size_t)2 * N];
      float b3 = bp[(size_t)3 * N];
      *reinterpret_cast<unsigned long long*>(&BsT[nn][kq]) =
          pack4bf(b0, b1, b2, b3);
    }
    if (k0 + BK < Kdim) {   // hint next K tile (global_prefetch)
      __builtin_prefetch(A + (size_t)(row0 + (tid >> 2)) * Kdim + k0 + BK, 0, 1);
      __builtin_prefetch(B + (size_t)(k0 + BK + (tid >> 2)) * N + col0, 0, 1);
    }
    __syncthreads();

#pragma unroll
    for (int kk = 0; kk < 2; ++kk) {
      FragBF af;
      const unsigned short* pa = &As[wm + m][kk * 32 + hsel];
      af.q[0] = *reinterpret_cast<const uint4v*>(pa);
      af.q[1] = *reinterpret_cast<const uint4v*>(pa + 8);
#pragma unroll
      for (int t = 0; t < 4; ++t) {
        FragBF bf;
        const unsigned short* pb = &BsT[wn + t * 16 + m][kk * 32 + hsel];
        bf.q[0] = *reinterpret_cast<const uint4v*>(pb);
        bf.q[1] = *reinterpret_cast<const uint4v*>(pb + 8);
        acc[t] = __builtin_amdgcn_wmma_f32_16x16x32_bf16(
            false, af.v, false, bf.v, (short)0, acc[t], false, false);
      }
    }
    __syncthreads();
  }

  const int cn = lane & 15;
  const int rofs = (lane < 16) ? 0 : 8;
#pragma unroll
  for (int t = 0; t < 4; ++t) {
#pragma unroll
    for (int r = 0; r < 8; ++r) {
      const int grow = row0 + wm + r + rofs;
      const size_t i0 = (size_t)grow * N + col0 + wn + t * 16 + cn;
      float v = acc[t][r];
      if (MODE == 2) {
        // single non-returning global_atomic_add_f32 (tiles disjoint; experts
        // serialize on the stream, so ordering is irrelevant)
        unsafeAtomicAdd(&C[i0], rowscale[(size_t)grow * rs_stride] * v);
      } else {
        if (MODE == 1) v += R[i0];
        C[i0] = v;
      }
    }
  }
}

// ---------------------------------------------------------------------------
// Per-head q/k RMSNorm + RoPE. One wave per (token, head).
// ---------------------------------------------------------------------------
__global__ void __launch_bounds__(32)
qk_norm_rope_kernel(float* __restrict__ q, const float* __restrict__ w, int nheads) {
  const int idx = blockIdx.x;
  const int tok = idx / nheads;
  const int hd  = idx % nheads;
  float* p = q + ((size_t)tok * nheads + hd) * HD_;
  const int lane = threadIdx.x;

  float x0 = p[lane],      x1 = p[lane + 32];
  float y0 = p[lane + 64], y1 = p[lane + 96];
  float ss = x0 * x0 + x1 * x1 + y0 * y0 + y1 * y1;
  ss = wave_reduce_sum(ss);
  const float r = rsqrtf(ss / (float)HD_ + EPS_);

  float a0 = x0 * r * w[lane],      a1 = x1 * r * w[lane + 32];
  float b0 = y0 * r * w[lane + 64], b1 = y1 * r * w[lane + 96];

  const float fs = (float)tok;
  float th0 = fs * __powf(10000.f, -2.f * (float)lane / (float)HD_);
  float th1 = fs * __powf(10000.f, -2.f * (float)(lane + 32) / (float)HD_);
  float c0 = cosf(th0), s0 = sinf(th0);
  float c1 = cosf(th1), s1 = sinf(th1);

  p[lane]      = a0 * c0 - b0 * s0;
  p[lane + 64] = b0 * c0 + a0 * s0;
  p[lane + 32] = a1 * c1 - b1 * s1;
  p[lane + 96] = b1 * c1 + a1 * s1;
}

// ---------------------------------------------------------------------------
// Flash attention (causal, GQA), WMMA bf16. One wave per (head, 16-q block).
// ---------------------------------------------------------------------------
__global__ void __launch_bounds__(32)
flash_attn_kernel(const float* __restrict__ Q, const float* __restrict__ Kg,
                  const float* __restrict__ Vg, float* __restrict__ O) {
  __shared__ unsigned short Qt[16][HD_ + PAD];    // row-major, K linear
  __shared__ unsigned short Kt[32][HD_ + PAD];    // row-major (B layout linear)
  __shared__ unsigned short VtT[HD_][32 + PAD];   // transposed: [hd][key]
  __shared__ unsigned short Pt[16][32 + PAD];     // row-major [q][key], K linear

  const int NQB  = T_ / 16;
  const int head = blockIdx.x / NQB;
  const int qb   = blockIdx.x % NQB;
  const int q0   = qb * 16;
  const int kvh  = head / (NH_ / NKV_);
  const int lane = threadIdx.x;
  const int n    = lane & 15;
  const int rofs = (lane < 16) ? 0 : 8;
  const int hsel = (lane < 16) ? 0 : 16;
  const float rscale = 0.08838834764831845f;   // HD^-0.5

  // stage Q tile (16x128), K linear
#pragma unroll
  for (int i = 0; i < 16; ++i) {
    int e = (lane + i * 32) * 4;
    int r = e >> 7, c = e & 127;
    float4 v = *reinterpret_cast<const float4*>(
        Q + ((size_t)(q0 + r) * NH_ + head) * HD_ + c);
    int cp = c;
    *reinterpret_cast<unsigned long long*>(&Qt[r][cp]) =
        pack4bf(v.x, v.y, v.z, v.w);
  }
  lds_fence();

  FragBF aq[4];
#pragma unroll
  for (int cc = 0; cc < 4; ++cc) {
    const unsigned short* pa = &Qt[n][cc * 32 + hsel];
    aq[cc].q[0] = *reinterpret_cast<const uint4v*>(pa);
    aq[cc].q[1] = *reinterpret_cast<const uint4v*>(pa + 8);
  }

  float m_i[8], l_i[8];
  v8f ov[8];
#pragma unroll
  for (int r = 0; r < 8; ++r) { m_i[r] = -1e30f; l_i[r] = 0.f; }
#pragma unroll
  for (int t = 0; t < 8; ++t) ov[t] = (v8f){0, 0, 0, 0, 0, 0, 0, 0};

  const int kend = q0 + 16;
  for (int kb0 = 0; kb0 < kend; kb0 += 32) {
    // stage K (row-major) and V (transposed) blocks, 32 keys x 128
#pragma unroll
    for (int i = 0; i < 32; ++i) {
      int e = (lane + i * 32) * 4;
      int r = e >> 7, c = e & 127;
      size_t gi = ((size_t)(kb0 + r) * NKV_ + kvh) * HD_ + c;
      float4 kv = *reinterpret_cast<const float4*>(Kg + gi);
      *reinterpret_cast<unsigned long long*>(&Kt[r][c]) =
          pack4bf(kv.x, kv.y, kv.z, kv.w);
      float4 vv = *reinterpret_cast<const float4*>(Vg + gi);
      VtT[c + 0][r] = f2bf(vv.x);
      VtT[c + 1][r] = f2bf(vv.y);
      VtT[c + 2][r] = f2bf(vv.z);
      VtT[c + 3][r] = f2bf(vv.w);
    }
    lds_fence();

    // scores: Q(16x128) x K^T(128x32) -> two 16x16 C tiles
    v8f c0 = (v8f){0, 0, 0, 0, 0, 0, 0, 0};
    v8f c1 = c0;
#pragma unroll
    for (int cc = 0; cc < 4; ++cc) {
      FragBF b0, b1;
      const unsigned short* p0p = &Kt[n][cc * 32 + hsel];
      const unsigned short* p1p = &Kt[16 + n][cc * 32 + hsel];
      b0.q[0] = *reinterpret_cast<const uint4v*>(p0p);
      b0.q[1] = *reinterpret_cast<const uint4v*>(p0p + 8);
      b1.q[0] = *reinterpret_cast<const uint4v*>(p1p);
      b1.q[1] = *reinterpret_cast<const uint4v*>(p1p + 8);
      c0 = __builtin_amdgcn_wmma_f32_16x16x32_bf16(false, aq[cc].v, false, b0.v,
                                                   (short)0, c0, false, false);
      c1 = __builtin_amdgcn_wmma_f32_16x16x32_bf16(false, aq[cc].v, false, b1.v,
                                                   (short)0, c1, false, false);
    }

    // online softmax (row = one 16-lane half of one C VGPR; permlane16 only)
#pragma unroll
    for (int r = 0; r < 8; ++r) {
      const int row = q0 + r + rofs;
      float s0 = c0[r] * rscale; if (kb0 + n      > row) s0 = -1e9f;
      float s1 = c1[r] * rscale; if (kb0 + 16 + n > row) s1 = -1e9f;
      float mx   = half_reduce_max(fmaxf(s0, s1));
      float mnew = fmaxf(m_i[r], mx);
      float scl  = __expf(m_i[r] - mnew);
      float p0 = __expf(s0 - mnew);
      float p1 = __expf(s1 - mnew);
      float rs = half_reduce_sum(p0 + p1);
      l_i[r] = l_i[r] * scl + rs;
      m_i[r] = mnew;
#pragma unroll
      for (int t = 0; t < 8; ++t) ov[t][r] *= scl;
      Pt[r + rofs][n]      = f2bf(p0);
      Pt[r + rofs][16 + n] = f2bf(p1);
    }
    lds_fence();

    // P (A fragment, K linear) x V (transposed, K linear)
    FragBF ap;
    {
      const unsigned short* pp = &Pt[n][hsel];
      ap.q[0] = *reinterpret_cast<const uint4v*>(pp);
      ap.q[1] = *reinterpret_cast<const uint4v*>(pp + 8);
    }
#pragma unroll
    for (int t = 0; t < 8; ++t) {
      FragBF bv;
      const unsigned short* pv = &VtT[t * 16 + n][hsel];
      bv.q[0] = *reinterpret_cast<const uint4v*>(pv);
      bv.q[1] = *reinterpret_cast<const uint4v*>(pv + 8);
      ov[t] = __builtin_amdgcn_wmma_f32_16x16x32_bf16(false, ap.v, false, bv.v,
                                                      (short)0, ov[t], false, false);
    }
    lds_fence();
  }

#pragma unroll
  for (int r = 0; r < 8; ++r) {
    const float inv = 1.f / l_i[r];
    const int row = q0 + r + rofs;
#pragma unroll
    for (int t = 0; t < 8; ++t)
      O[((size_t)row * NH_ + head) * HD_ + t * 16 + n] = ov[t][r] * inv;
  }
}

// ---------------------------------------------------------------------------
// Router: logits -> softmax -> top-8 -> dense combine[T,E]. One wave/token.
// ---------------------------------------------------------------------------
__global__ void __launch_bounds__(32)
router_kernel(const float* __restrict__ h2, const float* __restrict__ Wg,
              float* __restrict__ combine) {
  const int t = blockIdx.x;
  const int lane = threadIdx.x;     // = expert index
  const float* hr = h2 + (size_t)t * H_;
  float acc = 0.f;
  for (int k = 0; k < H_; k += 4) {
    acc += hr[k]     * Wg[(size_t)(k)     * E_ + lane];
    acc += hr[k + 1] * Wg[(size_t)(k + 1) * E_ + lane];
    acc += hr[k + 2] * Wg[(size_t)(k + 2) * E_ + lane];
    acc += hr[k + 3] * Wg[(size_t)(k + 3) * E_ + lane];
  }
  const float m = wave_reduce_max(acc);
  const float e = __expf(acc - m);
  const float s = wave_reduce_sum(e);
  const float prob = e / s;

  float p2 = prob, comb = 0.f;
#pragma unroll
  for (int i = 0; i < TOPK_; ++i) {
    float mx = wave_reduce_max(p2);
    unsigned long long msk = __ballot(p2 == mx);
    int win = __ffsll(msk) - 1;     // first occurrence, like top_k
    if (lane == win) { comb = p2; p2 = -1.f; }
  }
  combine[(size_t)t * E_ + lane] = comb;
}

// ---------------------------------------------------------------------------
// act[t,i] = silu(gu[t,i]) * gu[t, I+i]
// ---------------------------------------------------------------------------
__global__ void __launch_bounds__(256)
silu_mul_kernel(const float* __restrict__ gu, float* __restrict__ act) {
  const int i = blockIdx.x * 256 + threadIdx.x;
  if (i >= T_ * I_) return;
  const int t = i / I_, j = i % I_;
  const float g = gu[(size_t)t * (2 * I_) + j];
  const float u = gu[(size_t)t * (2 * I_) + I_ + j];
  act[i] = (g / (1.f + __expf(-g))) * u;
}

// ---------------------------------------------------------------------------
// Host orchestration
// ---------------------------------------------------------------------------
extern "C" void kernel_launch(void* const* d_in, const int* in_sizes, int n_in,
                              void* d_out, int out_size, void* d_ws, size_t ws_size,
                              hipStream_t stream) {
  (void)in_sizes; (void)n_in; (void)out_size; (void)ws_size;
  const float* x    = (const float*)d_in[0];
  const float* ln1w = (const float*)d_in[1];
  const float* ln2w = (const float*)d_in[2];
  const float* qnw  = (const float*)d_in[3];
  const float* knw  = (const float*)d_in[4];
  const float* Wq   = (const float*)d_in[5];
  const float* Wk   = (const float*)d_in[6];
  const float* Wv   = (const float*)d_in[7];
  const float* Wo   = (const float*)d_in[8];
  const float* Wg   = (const float*)d_in[9];
  const float* Wgu  = (const float*)d_in[10];
  const float* Wdn  = (const float*)d_in[11];
  float* out = (float*)d_out;

  float* ws  = (float*)d_ws;
  float* h1  = ws; ws += (size_t)T_ * H_;
  float* qb  = ws; ws += (size_t)T_ * NH_ * HD_;
  float* kb  = ws; ws += (size_t)T_ * NKV_ * HD_;
  float* vb  = ws; ws += (size_t)T_ * NKV_ * HD_;
  float* ao  = ws; ws += (size_t)T_ * NH_ * HD_;
  float* h2  = ws; ws += (size_t)T_ * H_;
  float* cmb = ws; ws += (size_t)T_ * E_;
  float* gu  = ws; ws += (size_t)T_ * 2 * I_;
  float* act = ws; ws += (size_t)T_ * I_;

  auto gemm = [&](const float* A, const float* Bm, float* C, const float* R,
                  const float* rs, int rs_stride, int M, int N, int K, int mode) {
    dim3 grid(N / BN, M / BM), block(256);
    if (mode == 0)
      gemm_bf16_kernel<0><<<grid, block, 0, stream>>>(A, Bm, C, R, rs, rs_stride, M, N, K);
    else if (mode == 1)
      gemm_bf16_kernel<1><<<grid, block, 0, stream>>>(A, Bm, C, R, rs, rs_stride, M, N, K);
    else
      gemm_bf16_kernel<2><<<grid, block, 0, stream>>>(A, Bm, C, R, rs, rs_stride, M, N, K);
  };

  rmsnorm_kernel<<<T_, 256, 0, stream>>>(x, ln1w, h1);

  gemm(h1, Wq, qb, nullptr, nullptr, 0, T_, NH_ * HD_,  H_, 0);
  gemm(h1, Wk, kb, nullptr, nullptr, 0, T_, NKV_ * HD_, H_, 0);
  gemm(h1, Wv, vb, nullptr, nullptr, 0, T_, NKV_ * HD_, H_, 0);

  qk_norm_rope_kernel<<<T_ * NH_,  32, 0, stream>>>(qb, qnw, NH_);
  qk_norm_rope_kernel<<<T_ * NKV_, 32, 0, stream>>>(kb, knw, NKV_);

  flash_attn_kernel<<<NH_ * (T_ / 16), 32, 0, stream>>>(qb, kb, vb, ao);

  gemm(ao, Wo, out, x, nullptr, 0, T_, H_, NH_ * HD_, 1);

  rmsnorm_kernel<<<T_, 256, 0, stream>>>(out, ln2w, h2);
  router_kernel<<<T_, 32, 0, stream>>>(h2, Wg, cmb);

  for (int e = 0; e < E_; ++e) {
    gemm(h2, Wgu + (size_t)e * H_ * 2 * I_, gu, nullptr, nullptr, 0,
         T_, 2 * I_, H_, 0);
    silu_mul_kernel<<<(T_ * I_ + 255) / 256, 256, 0, stream>>>(gu, act);
    gemm(act, Wdn + (size_t)e * I_ * H_, out, nullptr, cmb + e, E_,
         T_, H_, I_, 2);
  }
}
